// SFPxLSTM_63651415327297
// MI455X (gfx1250) — matrix operand order repaired and checked by reading, not compiled, not run on hardware
//
#include <hip/hip_runtime.h>
#include <hip/hip_bf16.h>
#include <math.h>

// ---------------------------------------------------------------------------
// SFPM forward for MI455X (gfx1250, wave32).
//  * Q/K/G are per-pixel affine in the 1-channel x -> never materialized.
//  * All 64x64 pointwise GEMMs: v_wmma_f32_16x16x32_bf16 (f32 accum).
//  * LDS layouts arranged so every WMMA fragment is contiguous per lane
//    -> ds_load_b128 instead of 16x ds_load_u16.
//  * Depthwise convs read neighbors from global (focal buffers are
//    L2-resident: 103MB < 192MB L2).
//  * top-k -> per-batch histogram threshold (2048 bins, one WG per batch).
// ---------------------------------------------------------------------------

typedef __attribute__((ext_vector_type(16))) __bf16 v16bf;
typedef __attribute__((ext_vector_type(8)))  __bf16 v8bf;
typedef __attribute__((ext_vector_type(8)))  float  v8f;

#define BATCH 8
#define HH    224
#define WW    224
#define HWPX  (HH*WW)          // 50176
#define HC    64
#define TOPK  15052            // max(int(0.3*224*224), 64)
#define TSB   72               // bf16 tile row stride (bank-conflict pad)
#define WSTR  72               // bf16 weight row stride

// prm layout (floats): see offsets below.
#define PRM_QA 0
#define PRM_QB 64
#define PRM_KA 128
#define PRM_KB 192
#define PRM_GA 256
#define PRM_GB 260
#define PRM_IA 264
#define PRM_IB 280
#define PRM_SB 296

// ---------------- WMMA fragment helpers (bf16 A/B, f32 C/D) ----------------
// A (16x32): lane L: M=L%16, half=L/16; elements a[i] (i=0..7)  = K ks*32+half*8+i
//                                       elements a[8+i]         = K ks*32+16+half*8+i
__device__ inline v16bf ldA(const __bf16* row /*[16][TSB]*/, int m, int half,
                            int ks) {
  const v8bf* p0 = (const v8bf*)(row + m * TSB + ks * 32 + half * 8);
  const v8bf* p1 = (const v8bf*)(row + m * TSB + ks * 32 + 16 + half * 8);
  v8bf lo = *p0, hi = *p1;
  return __builtin_shufflevector(lo, hi, 0, 1, 2, 3, 4, 5, 6, 7, 8, 9, 10, 11,
                                 12, 13, 14, 15);
}

// B (32x16): lane L: N=L%16 (+16*nt), half=L/16; elements b[i] = W[ks*32+half*8+i][n],
// b[8+i] = W[ks*32+16+half*8+i][n]
// weights staged as wT[n][k] (padded stride WSTR) -> two runs of 8 consecutive bf16.
__device__ inline v16bf ldB(const __bf16* w, int m, int half, int ks, int nt) {
  const __bf16* base = w + (m + nt * 16) * WSTR + ks * 32 + half * 8;
  const v8bf* p = (const v8bf*)base;
  v8bf lo = p[0], hi = p[2];
  return __builtin_shufflevector(lo, hi, 0, 1, 2, 3, 4, 5, 6, 7, 8, 9, 10, 11,
                                 12, 13, 14, 15);
}

__device__ inline void st8bf(__bf16* dst, const float* v) {  // 16B store
  v8bf o;
#pragma unroll
  for (int i = 0; i < 8; ++i) o[i] = (__bf16)v[i];
  *(v8bf*)dst = o;
}

__device__ inline v8f splat8(float v) {
  v8f r;
#pragma unroll
  for (int i = 0; i < 8; ++i) r[i] = v;
  return r;
}

// ---------------------------- prep: fold params ----------------------------
__global__ __launch_bounds__(256) void prep_kernel(
    const float* __restrict__ w_in, const float* __restrict__ b_in,
    const float* __restrict__ imp1_w, const float* __restrict__ imp1_b,
    const float* __restrict__ pw0_b, const float* __restrict__ bn0_g,
    const float* __restrict__ bn0_b, const float* __restrict__ bn0_m,
    const float* __restrict__ bn0_v, const float* __restrict__ pw1_b,
    const float* __restrict__ bn1_g, const float* __restrict__ bn1_b,
    const float* __restrict__ bn1_m, const float* __restrict__ bn1_v,
    const float* __restrict__ pw2_b, const float* __restrict__ bn2_g,
    const float* __restrict__ bn2_b, const float* __restrict__ bn2_m,
    const float* __restrict__ bn2_v, float* __restrict__ prm) {
  int t = threadIdx.x;
  if (t < 64) {
    prm[PRM_QA + t] = w_in[t];
    prm[PRM_QB + t] = b_in[t];
    prm[PRM_KA + t] = w_in[64 + t];
    prm[PRM_KB + t] = b_in[64 + t];
    const float* pb[3] = {pw0_b, pw1_b, pw2_b};
    const float* bg[3] = {bn0_g, bn1_g, bn2_g};
    const float* bb[3] = {bn0_b, bn1_b, bn2_b};
    const float* bm[3] = {bn0_m, bn1_m, bn2_m};
    const float* bv[3] = {bn0_v, bn1_v, bn2_v};
    for (int l = 0; l < 3; ++l) {
      float sc = bg[l][t] * rsqrtf(bv[l][t] + 1e-5f);
      prm[PRM_SB + l * 128 + t] = sc;
      prm[PRM_SB + l * 128 + 64 + t] = (pb[l][t] - bm[l][t]) * sc + bb[l][t];
    }
  }
  if (t < 4) {
    prm[PRM_GA + t] = w_in[128 + t];
    prm[PRM_GB + t] = b_in[128 + t];
  }
  if (t < 16) {
    float Aj = 0.f, Bj = 0.f;
    for (int c = 0; c < 64; ++c) {
      Aj += imp1_w[t * 64 + c] * w_in[64 + c];
      Bj += imp1_w[t * 64 + c] * b_in[64 + c];
    }
    prm[PRM_IA + t] = Aj;
    prm[PRM_IB + t] = Bj + imp1_b[t];
  }
}

// --------------------------- importance mask m -----------------------------
__global__ __launch_bounds__(256) void mask_kernel(
    const float* __restrict__ x, const float* __restrict__ prm,
    const float* __restrict__ imp2_w, const float* __restrict__ imp2_b,
    float* __restrict__ mbuf) {
  int gid = blockIdx.x * 256 + threadIdx.x;  // b*HW + p
  int b = gid / HWPX, p = gid - b * HWPX;
  int h = p / WW, w = p - h * WW;
  float Aj[16], Bj[16];
#pragma unroll
  for (int j = 0; j < 16; ++j) {
    Aj[j] = prm[PRM_IA + j];
    Bj[j] = prm[PRM_IB + j];
  }
  float s = imp2_b[0];
#pragma unroll
  for (int dy = -1; dy <= 1; ++dy) {
    int yy = h + dy;
    if (yy < 0 || yy >= HH) continue;
#pragma unroll
    for (int dx = -1; dx <= 1; ++dx) {
      int xx = w + dx;
      if (xx < 0 || xx >= WW) continue;
      float xn = x[b * HWPX + yy * WW + xx];
#pragma unroll
      for (int j = 0; j < 16; ++j) {
        float r = Aj[j] * xn + Bj[j];
        r = r > 0.f ? r : 0.f;
        s += r * imp2_w[j * 9 + (dy + 1) * 3 + (dx + 1)];
      }
    }
  }
  mbuf[gid] = 1.f / (1.f + __expf(-s));
}

// ------------------- per-batch top-k threshold (histogram) -----------------
#define NBINS 2048
__global__ __launch_bounds__(1024) void topk_kernel(
    const float* __restrict__ mbuf, float* __restrict__ thr) {
  __shared__ unsigned hist[NBINS];
  int b = blockIdx.x;
  for (int i = threadIdx.x; i < NBINS; i += 1024) hist[i] = 0u;
  __syncthreads();
  for (int i = threadIdx.x; i < HWPX; i += 1024) {
    float v = mbuf[b * HWPX + i];
    int bin = (int)(v * (float)NBINS);
    bin = bin < 0 ? 0 : (bin > NBINS - 1 ? NBINS - 1 : bin);
    atomicAdd(&hist[bin], 1u);
  }
  __syncthreads();
  if (threadIdx.x == 0) {
    unsigned c = 0;
    int bin = NBINS - 1;
    for (; bin > 0; --bin) {
      c += hist[bin];
      if (c >= (unsigned)TOPK) break;
    }
    thr[b] = (float)bin / (float)NBINS;  // mask = (m >= thr)
  }
}

// ------------- focal level: depthwise conv + WMMA pointwise+BN -------------
// Block: 256 thr = 8 waves; each wave owns 16 consecutive pixels.
template <int KSZ, int DIL, bool FIRST>
__global__ __launch_bounds__(256) void level_kernel(
    const float* __restrict__ in,    // focal_{l-1}; unused if FIRST
    const float* __restrict__ x,     // K = ka*x+kb when FIRST
    const float* __restrict__ mbuf, const float* __restrict__ thr,
    const float* __restrict__ prm, const float* __restrict__ dw_w,
    const float* __restrict__ pw_w, const float* __restrict__ sb,
    float* __restrict__ outF) {
  __shared__ __align__(16) __bf16 wB[64 * WSTR];
  __shared__ __align__(16) __bf16 tile[8][16 * TSB];
  const int t = threadIdx.x;
  const int wave = t >> 5, lane = t & 31;
  const int m = lane & 15, half = lane >> 4;
  // stage pointwise weights: wB[n*WSTR+k] = pw_w[n][k] (B[k][n])
  for (int i = t; i < 4096; i += 256) {
    int n = i >> 6, k = i & 63;
    wB[n * WSTR + k] = (__bf16)pw_w[i];
  }
  const int gp0 = blockIdx.x * 128;  // 128 | HWPX -> block stays in one image
  const int b = gp0 / HWPX;
  const int p0 = gp0 - b * HWPX;
  const int px = p0 + wave * 16 + m;
  const int h = px / WW, w = px - (px / WW) * WW;
  const float tb = FIRST ? thr[b] : 0.f;
  constexpr int KK = KSZ * KSZ;
  // depthwise: this lane computes 32 channels (4 chunks of 8) for its pixel
#pragma unroll 1
  for (int c8 = 0; c8 < 4; ++c8) {
    const int ch0 = half * 32 + c8 * 8;
    float acc8[8];
#pragma unroll
    for (int i = 0; i < 8; ++i) acc8[i] = 0.f;
#pragma unroll
    for (int ky = 0; ky < KSZ; ++ky) {
      int yy = h + (ky - KSZ / 2) * DIL;
      if (yy < 0 || yy >= HH) continue;
#pragma unroll
      for (int kx = 0; kx < KSZ; ++kx) {
        int xx = w + (kx - KSZ / 2) * DIL;
        if (xx < 0 || xx >= WW) continue;
        int pp = yy * WW + xx;
        if (FIRST) {
          float msk = (mbuf[b * HWPX + pp] >= tb) ? 1.f : 0.f;
          float mx = msk * x[b * HWPX + pp];
#pragma unroll
          for (int i = 0; i < 8; ++i) {
            int ch = ch0 + i;
            float kv = prm[PRM_KA + ch] * mx + msk * prm[PRM_KB + ch];
            acc8[i] += kv * dw_w[ch * KK + ky * KSZ + kx];
          }
        } else {
#pragma unroll
          for (int i = 0; i < 8; ++i) {
            int ch = ch0 + i;
            acc8[i] += in[((size_t)(b * 64 + ch)) * HWPX + pp] *
                       dw_w[ch * KK + ky * KSZ + kx];
          }
        }
      }
    }
    st8bf(&tile[wave][m * TSB + ch0], acc8);
  }
  __syncthreads();
  // pointwise 64->64 via 8x v_wmma_f32_16x16x32_bf16 per wave
  v8f acc[4];
#pragma unroll
  for (int nt = 0; nt < 4; ++nt) acc[nt] = splat8(0.f);
#pragma unroll
  for (int ks = 0; ks < 2; ++ks) {
    v16bf A = ldA(tile[wave], m, half, ks);
#pragma unroll
    for (int nt = 0; nt < 4; ++nt) {
      v16bf Bf = ldB(wB, m, half, ks, nt);
      acc[nt] = __builtin_amdgcn_wmma_f32_16x16x32_bf16(
          false, A, false, Bf, (short)0, acc[nt], false, false);
    }
  }
  // folded BN (+pw bias) + relu; D layout: N=m+nt*16, M=half*8+v
#pragma unroll
  for (int nt = 0; nt < 4; ++nt) {
    int n = m + nt * 16;
    float sc = sb[n], bt = sb[64 + n];
    float4 o0, o1;
    float* o = (float*)&o0;  // o0,o1 contiguous on stack? build separately:
    float tmp[8];
#pragma unroll
    for (int v = 0; v < 8; ++v) {
      float y = acc[nt][v] * sc + bt;
      tmp[v] = y > 0.f ? y : 0.f;
    }
    o0 = make_float4(tmp[0], tmp[1], tmp[2], tmp[3]);
    o1 = make_float4(tmp[4], tmp[5], tmp[6], tmp[7]);
    (void)o;
    size_t base = ((size_t)(b * 64 + n)) * HWPX + p0 + wave * 16 + half * 8;
    *(float4*)(outF + base) = o0;
    *(float4*)(outF + base + 4) = o1;
  }
}

// ---------------------- global mean of focal2 per (b,c) --------------------
__global__ __launch_bounds__(256) void gmean_kernel(
    const float* __restrict__ f2, float* __restrict__ kglob) {
  __shared__ float red[256];
  int bc = blockIdx.x;
  const float4* p = (const float4*)(f2 + (size_t)bc * HWPX);
  float s = 0.f;
  for (int i = threadIdx.x; i < HWPX / 4; i += 256) {
    float4 v = p[i];
    s += v.x + v.y + v.z + v.w;
  }
  red[threadIdx.x] = s;
  __syncthreads();
  for (int st = 128; st > 0; st >>= 1) {
    if (threadIdx.x < st) red[threadIdx.x] += red[threadIdx.x + st];
    __syncthreads();
  }
  if (threadIdx.x == 0) kglob[bc] = red[0] / (float)HWPX;
}

// ------- final fused: softmax gate, agg, V-GEMM, Q*V, out/fc GEMMs ---------
__global__ __launch_bounds__(256) void final_kernel(
    const float* __restrict__ x, const float* __restrict__ f0,
    const float* __restrict__ f1, const float* __restrict__ f2,
    const float* __restrict__ kglob, const float* __restrict__ prm,
    const float* __restrict__ v_w, const float* __restrict__ v_b,
    const float* __restrict__ out_w, const float* __restrict__ out_b,
    const float* __restrict__ fc_w, const float* __restrict__ fc_b,
    float* __restrict__ out) {
  __shared__ __align__(16) __bf16 wV[64 * WSTR], wO[64 * WSTR], wF[64 * WSTR];
  __shared__ __align__(16) __bf16 tile[8][16 * TSB];
  const int t = threadIdx.x;
  const int wave = t >> 5, lane = t & 31;
  const int m = lane & 15, half = lane >> 4;
  for (int i = t; i < 4096; i += 256) {
    int n = i >> 6, k = i & 63;
    wV[n * WSTR + k] = (__bf16)v_w[i];    // V[n]  = sum_k v_w[n][k]*Kf[k]
    wO[n * WSTR + k] = (__bf16)out_w[i];  // O[d]  = sum_c out_w[d][c]*QV[c]
    wF[n * WSTR + k] = (__bf16)fc_w[i];   // P[n]  = sum_k fc_w[n][k]*Q[k]
  }
  const int gp0 = blockIdx.x * 128;
  const int b = gp0 / HWPX;
  const int p0 = gp0 - b * HWPX;
  const int pxl = p0 + wave * 16 + m;
  const float xv = x[b * HWPX + pxl];
  // softmax gate for this lane's pixel
  float g[4], mx = -1e30f;
#pragma unroll
  for (int j = 0; j < 4; ++j) {
    g[j] = prm[PRM_GA + j] * xv + prm[PRM_GB + j];
    mx = g[j] > mx ? g[j] : mx;
  }
  float se = 0.f;
#pragma unroll
  for (int j = 0; j < 4; ++j) {
    g[j] = __expf(g[j] - mx);
    se += g[j];
  }
  float inv = 1.f / se;
#pragma unroll
  for (int j = 0; j < 4; ++j) g[j] *= inv;
  // K_final into LDS [px][ch] (bf16, 8-channel packed stores)
#pragma unroll 1
  for (int c8 = 0; c8 < 4; ++c8) {
    int ch0 = half * 32 + c8 * 8;
    float kf[8];
#pragma unroll
    for (int i = 0; i < 8; ++i) {
      int ch = ch0 + i;
      size_t idx = ((size_t)(b * 64 + ch)) * HWPX + pxl;
      kf[i] = g[0] * f0[idx] + g[1] * f1[idx] + g[2] * f2[idx] +
              g[3] * kglob[b * 64 + ch];
    }
    st8bf(&tile[wave][m * TSB + ch0], kf);
  }
  __syncthreads();
  // V = Kf x v_w + v_b
  v8f aV[4];
#pragma unroll
  for (int nt = 0; nt < 4; ++nt) aV[nt] = splat8(v_b[m + nt * 16]);
#pragma unroll
  for (int ks = 0; ks < 2; ++ks) {
    v16bf A = ldA(tile[wave], m, half, ks);
#pragma unroll
    for (int nt = 0; nt < 4; ++nt) {
      v16bf Bf = ldB(wV, m, half, ks, nt);
      aV[nt] = __builtin_amdgcn_wmma_f32_16x16x32_bf16(
          false, A, false, Bf, (short)0, aV[nt], false, false);
    }
  }
  __syncthreads();
  // QV = Q .* V in D layout, restage to LDS [px][ch]
  float xm[8];
#pragma unroll
  for (int v = 0; v < 8; ++v)
    xm[v] = x[b * HWPX + p0 + wave * 16 + half * 8 + v];
#pragma unroll
  for (int nt = 0; nt < 4; ++nt) {
    int n = m + nt * 16;
    float qa = prm[PRM_QA + n], qb = prm[PRM_QB + n];
#pragma unroll
    for (int v = 0; v < 8; ++v) {
      float q = qa * xm[v] + qb;
      tile[wave][(half * 8 + v) * TSB + n] = (__bf16)(q * aV[nt][v]);
    }
  }
  __syncthreads();
  // O = QV x out_w^T + out_b  ;  x_proj = Q x fc_w^T + fc_b
  v8f aO[4], aP[4];
#pragma unroll
  for (int nt = 0; nt < 4; ++nt) {
    aO[nt] = splat8(out_b[m + nt * 16]);
    aP[nt] = splat8(fc_b[m + nt * 16]);
  }
#pragma unroll
  for (int ks = 0; ks < 2; ++ks) {
    v16bf Aqv = ldA(tile[wave], m, half, ks);
    v16bf Aq;  // Q rebuilt in registers: A layout M=m (pixel), K=channel
#pragma unroll
    for (int i = 0; i < 8; ++i) {
      int k0 = ks * 32 + half * 8 + i;        // elements 0..7
      int k1 = ks * 32 + 16 + half * 8 + i;   // elements 8..15
      Aq[i]     = (__bf16)(prm[PRM_QA + k0] * xv + prm[PRM_QB + k0]);
      Aq[8 + i] = (__bf16)(prm[PRM_QA + k1] * xv + prm[PRM_QB + k1]);
    }
#pragma unroll
    for (int nt = 0; nt < 4; ++nt) {
      v16bf Bo = ldB(wO, m, half, ks, nt);
      aO[nt] = __builtin_amdgcn_wmma_f32_16x16x32_bf16(
          false, Aqv, false, Bo, (short)0, aO[nt], false, false);
      v16bf Bp = ldB(wF, m, half, ks, nt);
      aP[nt] = __builtin_amdgcn_wmma_f32_16x16x32_bf16(
          false, Aq, false, Bp, (short)0, aP[nt], false, false);
    }
  }
#pragma unroll
  for (int nt = 0; nt < 4; ++nt) {
    int n = m + nt * 16;
    float tmp[8];
#pragma unroll
    for (int v = 0; v < 8; ++v) tmp[v] = aO[nt][v] + aP[nt][v];
    size_t base = ((size_t)(b * 64 + n)) * HWPX + p0 + wave * 16 + half * 8;
    *(float4*)(out + base) = make_float4(tmp[0], tmp[1], tmp[2], tmp[3]);
    *(float4*)(out + base + 4) = make_float4(tmp[4], tmp[5], tmp[6], tmp[7]);
  }
}

// ------------------------------- launcher ----------------------------------
extern "C" void kernel_launch(void* const* d_in, const int* in_sizes, int n_in,
                              void* d_out, int out_size, void* d_ws,
                              size_t ws_size, hipStream_t stream) {
  const float* x      = (const float*)d_in[0];
  const float* w_in   = (const float*)d_in[1];
  const float* b_in   = (const float*)d_in[2];
  const float* dw0_w  = (const float*)d_in[3];
  const float* pw0_w  = (const float*)d_in[4];
  const float* pw0_b  = (const float*)d_in[5];
  const float* bn0_g  = (const float*)d_in[6];
  const float* bn0_b  = (const float*)d_in[7];
  const float* bn0_m  = (const float*)d_in[8];
  const float* bn0_v  = (const float*)d_in[9];
  const float* dw1_w  = (const float*)d_in[10];
  const float* pw1_w  = (const float*)d_in[11];
  const float* pw1_b  = (const float*)d_in[12];
  const float* bn1_g  = (const float*)d_in[13];
  const float* bn1_b  = (const float*)d_in[14];
  const float* bn1_m  = (const float*)d_in[15];
  const float* bn1_v  = (const float*)d_in[16];
  const float* dw2_w  = (const float*)d_in[17];
  const float* pw2_w  = (const float*)d_in[18];
  const float* pw2_b  = (const float*)d_in[19];
  const float* bn2_g  = (const float*)d_in[20];
  const float* bn2_b  = (const float*)d_in[21];
  const float* bn2_m  = (const float*)d_in[22];
  const float* bn2_v  = (const float*)d_in[23];
  const float* imp1_w = (const float*)d_in[24];
  const float* imp1_b = (const float*)d_in[25];
  const float* imp2_w = (const float*)d_in[26];
  const float* imp2_b = (const float*)d_in[27];
  const float* v_w    = (const float*)d_in[28];
  const float* v_b    = (const float*)d_in[29];
  const float* out_w  = (const float*)d_in[30];
  const float* out_b  = (const float*)d_in[31];
  const float* fc_w   = (const float*)d_in[32];
  const float* fc_b   = (const float*)d_in[33];
  float* out = (float*)d_out;

  // workspace carve-up
  float* ws = (float*)d_ws;
  size_t o = 0;
  float* prm   = ws + o; o += 1024;
  float* thr   = ws + o; o += 32;
  float* kglob = ws + o; o += 512;
  float* mbuf  = ws + o; o += (size_t)BATCH * HWPX;
  float* f0    = ws + o; o += (size_t)BATCH * 64 * HWPX;
  float* f1    = ws + o; o += (size_t)BATCH * 64 * HWPX;
  float* f2    = ws + o; o += (size_t)BATCH * 64 * HWPX;
  (void)o; (void)ws_size; (void)in_sizes; (void)n_in; (void)out_size;

  const int nPix = BATCH * HWPX;         // 401408
  const int gemmBlocks = nPix / 128;     // 3136
  const int maskBlocks = nPix / 256;     // 1568

  prep_kernel<<<1, 256, 0, stream>>>(w_in, b_in, imp1_w, imp1_b, pw0_b, bn0_g,
                                     bn0_b, bn0_m, bn0_v, pw1_b, bn1_g, bn1_b,
                                     bn1_m, bn1_v, pw2_b, bn2_g, bn2_b, bn2_m,
                                     bn2_v, prm);
  mask_kernel<<<maskBlocks, 256, 0, stream>>>(x, prm, imp2_w, imp2_b, mbuf);
  topk_kernel<<<BATCH, 1024, 0, stream>>>(mbuf, thr);
  level_kernel<3, 1, true><<<gemmBlocks, 256, 0, stream>>>(
      nullptr, x, mbuf, thr, prm, dw0_w, pw0_w, prm + PRM_SB + 0 * 128, f0);
  level_kernel<5, 2, false><<<gemmBlocks, 256, 0, stream>>>(
      f0, x, mbuf, thr, prm, dw1_w, pw1_w, prm + PRM_SB + 1 * 128, f1);
  level_kernel<7, 4, false><<<gemmBlocks, 256, 0, stream>>>(
      f1, x, mbuf, thr, prm, dw2_w, pw2_w, prm + PRM_SB + 2 * 128, f2);
  gmean_kernel<<<BATCH * 64, 256, 0, stream>>>(f2, kglob);
  final_kernel<<<gemmBlocks, 256, 0, stream>>>(x, f0, f1, f2, kglob, prm, v_w,
                                               v_b, out_w, out_b, fc_w, fc_b,
                                               out);
}
